// MultiHeadQKVCreator_80118319940349
// MI455X (gfx1250) — compile-verified, matrix operand order checked by reading, not run on hardware
//
#include <hip/hip_runtime.h>

typedef __bf16 bf16_t;
typedef __attribute__((ext_vector_type(16))) __bf16 v16bf;
typedef __attribute__((ext_vector_type(8)))  __bf16 v8bf;
typedef __attribute__((ext_vector_type(4)))  __bf16 v4bf;
typedef __attribute__((ext_vector_type(8)))  float  v8f;
typedef unsigned int u32x4 __attribute__((ext_vector_type(4)));
typedef int          i32x8 __attribute__((ext_vector_type(8)));
typedef int          i32x4 __attribute__((ext_vector_type(4)));

namespace {
constexpr int kHidden  = 4096;
constexpr int kHeads   = 32;
constexpr int kHeadDim = 128;
constexpr int kSeq     = 2048;
constexpr int kCache   = 2048;
constexpr int kBatch   = 4;
constexpr int kM       = kBatch * kSeq;   // 8192 rows of x
constexpr int TM = 128;                   // M tile
constexpr int TN = 128;                   // N tile == one head
constexpr int TK = 32;                    // K chunk (bf16 WMMA depth)
constexpr int NKT = kHidden / TK;         // 128 K iterations
constexpr int kOutStride = 132;           // padded LDS row stride (floats)
constexpr float kLnBaseOver64 = 0.14391156831f;  // ln(10000)/64
}

struct Smem {
  union {
    struct {
      bf16_t A[2][TM * TK];   // 2 x 8 KB
      bf16_t B[2][TN * TK];   // 2 x 8 KB
    } g;
    float out[64 * kOutStride];  // 33.8 KB epilogue staging (two 64-row passes)
  };
};

// ---------------------------------------------------------------------------
// TDM: issue one 2D tensor_load_to_lds of a TM x TK bf16 tile whose rows are
// kHidden bf16 apart in global memory. D# layout per CDNA5 ISA §8.3/8.4.
// data_size = 8B units: row = TK*2/8 = 8 units, row stride = kHidden*2/8.
// ---------------------------------------------------------------------------
__device__ __forceinline__ void tdm_load_tile(unsigned int lds_off,
                                              const void* gptr) {
  const unsigned long long ga = (unsigned long long)(uintptr_t)gptr;
  const unsigned int td0   = (kHidden * 2) / 8;  // tensor_dim0 & stride: 1024
  const unsigned int tile0 = (TK * 2) / 8;       // 8
  const unsigned int tile1 = TM;                 // 128
  const unsigned int trows = TM;                 // rows available from tile org

  u32x4 g0;
  g0[0] = 1u;                                        // count=1, user mode
  g0[1] = lds_off;                                   // lds_addr
  g0[2] = (unsigned int)ga;                          // global_addr[31:0]
  g0[3] = ((unsigned int)(ga >> 32) & 0x01FFFFFFu)   // global_addr[56:32]
          | (2u << 30);                              // type=2 ("image")

  i32x8 g1;
  g1[0] = (int)(3u << 16);                           // data_size=3 (8B units)
  g1[1] = (int)((td0 & 0xFFFFu) << 16);              // tensor_dim0[15:0]
  g1[2] = (int)(((td0 >> 16) & 0xFFFFu) |            // tensor_dim0[31:16]
                ((trows & 0xFFFFu) << 16));          // tensor_dim1[15:0]
  g1[3] = (int)(((trows >> 16) & 0xFFFFu) |          // tensor_dim1[31:16]
                ((tile0 & 0xFFFFu) << 16));          // tile_dim0
  g1[4] = (int)(tile1 & 0xFFFFu);                    // tile_dim1 (tile_dim2=0)
  g1[5] = (int)td0;                                  // tensor_dim0_stride[31:0]
  g1[6] = 0;                                         // stride hi / dim1_stride lo
  g1[7] = 0;

  const i32x4 z4 = {0, 0, 0, 0};
#if __has_include(<hip/amd_detail/amd_gfx1250_TDM.h>)
  const i32x8 z8 = {0, 0, 0, 0, 0, 0, 0, 0};
  __builtin_amdgcn_tensor_load_to_lds(g0, g1, z4, z4, z8, 0);   // clang-23 form
#else
  __builtin_amdgcn_tensor_load_to_lds(g0, g1, z4, z4, 0);       // ROCm 7.2 form
#endif
}

// ---------------------------------------------------------------------------
// Load one 16x32 bf16 fragment from LDS laid out row-major, K contiguous
// (row stride = TK elements = 64 B). Matches the ISA 16-bit A/B VGPR layout:
// lanes 0-15 -> K {0..7, 16..23}, lanes 16-31 -> K {8..15, 24..31}.
// ---------------------------------------------------------------------------
__device__ __forceinline__ v16bf load_frag(const bf16_t* base, int lane) {
  const int r  = lane & 15;
  const int kq = lane >> 4;  // 0 or 1
  union { v16bf v; v8bf h[2]; } u;
  u.h[0] = *reinterpret_cast<const v8bf*>(base + r * TK + kq * 8);
  u.h[1] = *reinterpret_cast<const v8bf*>(base + r * TK + 16 + kq * 8);
  return u.v;
}

// ---------------------------------------------------------------------------
// Fallback fill: convert one K-chunk of X (A: 128x32) and W (B: 128x32,
// N-major) from f32 global to bf16 LDS. All 8 global b128 loads are issued
// before any conversion so loadcnt waits amortize.
// ---------------------------------------------------------------------------
__device__ __forceinline__ void fill_stage(const float* __restrict__ X,
                                           const float* __restrict__ W,
                                           bf16_t* __restrict__ As,
                                           bf16_t* __restrict__ Bs,
                                           int m0, int e0, int k0, int tid) {
  float4 fa[4], fb[4];
#pragma unroll
  for (int i = 0; i < 4; ++i) {
    const int slot = tid + (i << 8);        // 0..1023
    const int row  = slot >> 3;             // 0..127
    const int kk   = (slot & 7) << 2;       // 0,4,..,28
    fa[i] = *reinterpret_cast<const float4*>(X + (size_t)(m0 + row) * kHidden + k0 + kk);
    fb[i] = *reinterpret_cast<const float4*>(W + (size_t)(e0 + row) * kHidden + k0 + kk);
  }
  // Speculative prefetch of the K-chunk after next (global_prefetch_b8).
  if (k0 + 2 * TK < kHidden) {
    const int row = tid & 127;
    __builtin_prefetch(X + (size_t)(m0 + row) * kHidden + k0 + 2 * TK, 0, 1);
    __builtin_prefetch(W + (size_t)(e0 + row) * kHidden + k0 + 2 * TK, 0, 1);
  }
#pragma unroll
  for (int i = 0; i < 4; ++i) {
    const int slot = tid + (i << 8);
    const int row  = slot >> 3;
    const int kk   = (slot & 7) << 2;
    v4bf pa, pb;
    pa[0] = (bf16_t)fa[i].x; pa[1] = (bf16_t)fa[i].y;
    pa[2] = (bf16_t)fa[i].z; pa[3] = (bf16_t)fa[i].w;
    pb[0] = (bf16_t)fb[i].x; pb[1] = (bf16_t)fb[i].y;
    pb[2] = (bf16_t)fb[i].z; pb[3] = (bf16_t)fb[i].w;
    *reinterpret_cast<v4bf*>(As + row * TK + kk) = pa;
    *reinterpret_cast<v4bf*>(Bs + row * TK + kk) = pb;
  }
}

// ---------------------------------------------------------------------------
// D = X * W^T + b for q/k/v (z = 0/1/2), fused RoPE + transpose epilogue.
// kTDM=true : operands are pre-converted bf16 mirrors; staging via TDM.
// kTDM=false: f32 operands converted in-kernel (no workspace needed).
// ---------------------------------------------------------------------------
template <bool kTDM>
__global__ __launch_bounds__(256)
void qkv_gemm_kernel(const float* __restrict__ x,
                     const float* __restrict__ wq, const float* __restrict__ wk,
                     const float* __restrict__ wv,
                     const bf16_t* __restrict__ xb,
                     const bf16_t* __restrict__ wqb,
                     const bf16_t* __restrict__ wkb,
                     const bf16_t* __restrict__ wvb,
                     const float* __restrict__ bq, const float* __restrict__ bk,
                     const float* __restrict__ bv,
                     float* __restrict__ qout, float* __restrict__ kout,
                     float* __restrict__ vout) {
  __shared__ Smem sm;
  const int tid  = threadIdx.x;
  const int lane = tid & 31;
  const int wid  = tid >> 5;     // 8 waves
  const int wm   = wid & 3;      // 4 waves along M (32 rows each)
  const int wn   = wid >> 2;     // 2 waves along N (64 cols each)

  const int m0 = blockIdx.x * TM;       // row tile in flattened (B*S)
  const int h  = blockIdx.y;            // head index; N tile == head
  const int e0 = h * kHeadDim;
  const int z  = blockIdx.z;            // 0=q 1=k 2=v

  const float*  W    = (z == 0) ? wq  : (z == 1) ? wk  : wv;
  const bf16_t* Wb   = (z == 0) ? wqb : (z == 1) ? wkb : wvb;
  const float*  bias = (z == 0) ? bq  : (z == 1) ? bk  : bv;

  v8f acc[2][4];
#pragma unroll
  for (int fr = 0; fr < 2; ++fr)
#pragma unroll
    for (int fc = 0; fc < 4; ++fc)
      acc[fr][fc] = (v8f){0.f, 0.f, 0.f, 0.f, 0.f, 0.f, 0.f, 0.f};

  unsigned int ldsA[2], ldsB[2];
  if constexpr (kTDM) {
    ldsA[0] = (unsigned int)(uintptr_t)&sm.g.A[0][0];
    ldsA[1] = (unsigned int)(uintptr_t)&sm.g.A[1][0];
    ldsB[0] = (unsigned int)(uintptr_t)&sm.g.B[0][0];
    ldsB[1] = (unsigned int)(uintptr_t)&sm.g.B[1][0];
    if (wid == 0) {
      tdm_load_tile(ldsA[0], xb + (size_t)m0 * kHidden);
      tdm_load_tile(ldsB[0], Wb + (size_t)e0 * kHidden);
      __builtin_amdgcn_s_wait_tensorcnt(0);
    }
  } else {
    fill_stage(x, W, sm.g.A[0], sm.g.B[0], m0, e0, 0, tid);
  }
  __syncthreads();

  int buf = 0;
  for (int kt = 0; kt < NKT; ++kt) {
    if (kt + 1 < NKT) {
      if constexpr (kTDM) {
        if (wid == 0) {
          const size_t koff = (size_t)(kt + 1) * TK;
          tdm_load_tile(ldsA[buf ^ 1], xb + (size_t)m0 * kHidden + koff);
          tdm_load_tile(ldsB[buf ^ 1], Wb + (size_t)e0 * kHidden + koff);
        }
      } else {
        fill_stage(x, W, sm.g.A[buf ^ 1], sm.g.B[buf ^ 1], m0, e0,
                   (kt + 1) * TK, tid);
      }
    }

    const bf16_t* Abase = sm.g.A[buf] + (wm * 32) * TK;
    const bf16_t* Bbase = sm.g.B[buf] + (wn * 64) * TK;
    v16bf af[2], bfm[4];
#pragma unroll
    for (int fr = 0; fr < 2; ++fr) af[fr] = load_frag(Abase + fr * 16 * TK, lane);
#pragma unroll
    for (int fc = 0; fc < 4; ++fc) bfm[fc] = load_frag(Bbase + fc * 16 * TK, lane);

#pragma unroll
    for (int fr = 0; fr < 2; ++fr)
#pragma unroll
      for (int fc = 0; fc < 4; ++fc)
        acc[fr][fc] = __builtin_amdgcn_wmma_f32_16x16x32_bf16(
            false, af[fr], false, bfm[fc], (short)0, acc[fr][fc], false, false);

    if constexpr (kTDM) {
      if (wid == 0) __builtin_amdgcn_s_wait_tensorcnt(0);
    }
    __syncthreads();
    buf ^= 1;
  }

  // ---- Epilogue: bias + RoPE + transposed coalesced write, via LDS. ----
  // C/D layout: VGPR r -> M = r + 8*(lane>=16), N = lane&15 (within 16x16).
  const int moff = (lane >> 4) << 3;
  const int ncol = lane & 15;

  float* O;
  int    rows, roffBase;
  bool   doRope;
  if (z == 0)      { O = qout; rows = kSeq;          roffBase = 0;      doRope = true;  }
  else if (z == 1) { O = kout; rows = kCache + kSeq; roffBase = kCache; doRope = true;  }
  else             { O = vout; rows = kCache + kSeq; roffBase = kCache; doRope = false; }

#pragma unroll
  for (int pass = 0; pass < 2; ++pass) {
    __syncthreads();
    if ((wm >> 1) == pass) {
      const int rbase = (wm & 1) * 32;   // local row within this 64-row pass
#pragma unroll
      for (int fr = 0; fr < 2; ++fr)
#pragma unroll
        for (int fc = 0; fc < 4; ++fc) {
          union { v8f v; float f[8]; } u;
          u.v = acc[fr][fc];
          const int ml = rbase + fr * 16 + moff;
          const int nl = wn * 64 + fc * 16 + ncol;
#pragma unroll
          for (int r = 0; r < 8; ++r)
            sm.out[(ml + r) * kOutStride + nl] = u.f[r];
        }
    }
    __syncthreads();

    // 64 rows x 32 float4 chunks = 2048 chunks -> 8 per thread.
#pragma unroll
    for (int i = 0; i < 8; ++i) {
      const int chunk = tid + (i << 8);
      const int rloc  = chunk >> 5;           // 0..63
      const int c4    = (chunk & 31) << 2;    // 0..124, step 4
      const int m     = m0 + pass * 64 + rloc;
      const int b     = m >> 11;              // / kSeq
      const int srow  = m & (kSeq - 1);

      const float* src = sm.out + rloc * kOutStride;
      float4 val = *reinterpret_cast<const float4*>(src + c4);
      float4 par = *reinterpret_cast<const float4*>(src + (c4 ^ 64));
      const float4 bv4 = *reinterpret_cast<const float4*>(bias + e0 + c4);
      const float4 bp4 = *reinterpret_cast<const float4*>(bias + e0 + (c4 ^ 64));
      val.x += bv4.x; val.y += bv4.y; val.z += bv4.z; val.w += bv4.w;
      par.x += bp4.x; par.y += bp4.y; par.z += bp4.z; par.w += bp4.w;

      float4 res = val;
      if (doRope) {
        const float pos = (float)(roffBase + srow);
        const int   p   = c4 & 63;
        float* vp = &val.x;
        float* pp = &par.x;
        float* rp = &res.x;
#pragma unroll
        for (int j = 0; j < 4; ++j) {
          const float ang = pos * __expf(-(float)(p + j) * kLnBaseOver64);
          float sn, cn;
          __sincosf(ang, &sn, &cn);
          rp[j] = (c4 < 64) ? (vp[j] * cn - pp[j] * sn)
                            : (pp[j] * sn + vp[j] * cn);
        }
      }

      const size_t off =
          ((size_t)(b * kHeads + h) * rows + (roffBase + srow)) * kHeadDim + c4;
      *reinterpret_cast<float4*>(O + off) = res;
    }
  }
}

// f32 -> bf16 mirror for the TDM path (pure bandwidth, v_cvt_pk_bf16_f32).
__global__ __launch_bounds__(256)
void cvt_bf16_kernel(const float* __restrict__ src, bf16_t* __restrict__ dst,
                     long n4) {
  const long stride = (long)gridDim.x * blockDim.x;
  for (long i = (long)blockIdx.x * blockDim.x + threadIdx.x; i < n4;
       i += stride) {
    const float4 f = reinterpret_cast<const float4*>(src)[i];
    v4bf p;
    p[0] = (bf16_t)f.x; p[1] = (bf16_t)f.y;
    p[2] = (bf16_t)f.z; p[3] = (bf16_t)f.w;
    reinterpret_cast<v4bf*>(dst)[i] = p;
  }
}

// Pure-bandwidth concat: cache (B,CL,H,D) -> out (B,H,CL+S,D) rows [0,CL).
__global__ __launch_bounds__(256)
void cache_copy_kernel(const float* __restrict__ ck, const float* __restrict__ cv,
                       float* __restrict__ kout, float* __restrict__ vout) {
  const long total4 = (long)kBatch * kCache * kHeads * (kHeadDim / 4);  // 8,388,608
  const long stride = (long)gridDim.x * blockDim.x;
  for (long idx = (long)blockIdx.x * blockDim.x + threadIdx.x; idx < total4;
       idx += stride) {
    const int  dd4 = (int)(idx & 31);
    const int  h   = (int)((idx >> 5) & 31);
    const long cl  = (idx >> 10) & (kCache - 1);
    const long b   = idx >> 21;
    const float4 kq = reinterpret_cast<const float4*>(ck)[idx];
    const float4 vq = reinterpret_cast<const float4*>(cv)[idx];
    const long o = (((b * kHeads + h) * (long)(kCache + kSeq)) + cl) *
                       (kHeadDim / 4) + dd4;
    reinterpret_cast<float4*>(kout)[o] = kq;
    reinterpret_cast<float4*>(vout)[o] = vq;
  }
}

extern "C" void kernel_launch(void* const* d_in, const int* in_sizes, int n_in,
                              void* d_out, int out_size, void* d_ws, size_t ws_size,
                              hipStream_t stream) {
  (void)in_sizes; (void)n_in; (void)out_size;
  const float* x  = (const float*)d_in[0];
  const float* ck = (const float*)d_in[1];
  const float* cv = (const float*)d_in[2];
  const float* wq = (const float*)d_in[3];
  const float* bq = (const float*)d_in[4];
  const float* wk = (const float*)d_in[5];
  const float* bk = (const float*)d_in[6];
  const float* wv = (const float*)d_in[7];
  const float* bv = (const float*)d_in[8];

  float* out  = (float*)d_out;
  float* qout = out;
  float* kout = qout + (size_t)kBatch * kHeads * kSeq * kHeadDim;
  float* vout = kout + (size_t)kBatch * kHeads * (kCache + kSeq) * kHeadDim;

  const size_t xElems = (size_t)kM * kHidden;
  const size_t wElems = (size_t)kHidden * kHidden;
  const size_t need   = (xElems + 3 * wElems) * sizeof(bf16_t);

  const dim3 grid(kM / TM, kHeads, 3);   // 64 x 32 x 3

  if (ws_size >= need) {
    // bf16 mirrors in workspace, GEMM staged through the Tensor Data Mover.
    bf16_t* xb  = (bf16_t*)d_ws;
    bf16_t* wqb = xb + xElems;
    bf16_t* wkb = wqb + wElems;
    bf16_t* wvb = wkb + wElems;
    cvt_bf16_kernel<<<dim3(2048), dim3(256), 0, stream>>>(x,  xb,  (long)(xElems / 4));
    cvt_bf16_kernel<<<dim3(1024), dim3(256), 0, stream>>>(wq, wqb, (long)(wElems / 4));
    cvt_bf16_kernel<<<dim3(1024), dim3(256), 0, stream>>>(wk, wkb, (long)(wElems / 4));
    cvt_bf16_kernel<<<dim3(1024), dim3(256), 0, stream>>>(wv, wvb, (long)(wElems / 4));
    qkv_gemm_kernel<true><<<grid, dim3(256), 0, stream>>>(
        x, wq, wk, wv, xb, wqb, wkb, wvb, bq, bk, bv, qout, kout, vout);
  } else {
    qkv_gemm_kernel<false><<<grid, dim3(256), 0, stream>>>(
        x, wq, wk, wv, nullptr, nullptr, nullptr, nullptr, bq, bk, bv,
        qout, kout, vout);
  }
  cache_copy_kernel<<<dim3(2048), dim3(256), 0, stream>>>(ck, cv, kout, vout);
}
